// MakeMoE_57750130262447
// MI455X (gfx1250) — matrix operand order repaired and checked by reading, not compiled, not run on hardware
//
#include <hip/hip_runtime.h>
#include <stdint.h>

#define B_  4096
#define D_  256
#define H_  1024
#define E_  8

typedef unsigned short u16;
typedef __attribute__((ext_vector_type(16))) __bf16 bf16x16;
typedef __attribute__((ext_vector_type(8)))  float  f32x8;

#if defined(__has_builtin)
#if __has_builtin(__builtin_amdgcn_sched_barrier)
#define SCHED_FENCE() __builtin_amdgcn_sched_barrier(0)
#endif
#endif
#ifndef SCHED_FENCE
#define SCHED_FENCE()
#endif

// LDS strides padded so A-fragment rows land on distinct banks
#define XS_STRIDE 264                       // 256 + 8
#define HS_STRIDE 1032                      // 1024 + 8

__device__ __forceinline__ u16 f2bf(float f) {   // f32 -> bf16 (RNE via native cvt)
  union { __bf16 b; u16 u; } v;
  v.b = (__bf16)f;
  return v.u;
}

__device__ __forceinline__ bf16x16 ld_frag(const u16* p0, const u16* p1) {
  union { uint4 q[2]; bf16x16 v; } u;
  u.q[0] = *(const uint4*)p0;   // 16B load (ds_load_b128 / global_load_b128)
  u.q[1] = *(const uint4*)p1;
  return u.v;
}

__global__ void k_zero(int* counts) {
  if (threadIdx.x < E_) counts[threadIdx.x] = 0;
}

__global__ void k_scatter(const int* __restrict__ eidx, int* counts, int* rows) {
  int t = blockIdx.x * blockDim.x + threadIdx.x;
  if (t < B_) {
    int e = eidx[t];
    int p = atomicAdd(&counts[e], 1);
    rows[e * B_ + p] = t;
  }
}

// src: [E][R][C] f32  ->  dst: [E][C][R] bf16   (32x32 LDS tile transpose, 256 thr)
__global__ void k_transpose_cvt(const float* __restrict__ src, u16* __restrict__ dst,
                                int R, int C) {
  __shared__ float tile[32][33];
  const int tilesC = C >> 5;
  const int tpe    = tilesC * (R >> 5);
  const int e  = blockIdx.x / tpe;
  const int bt = blockIdx.x % tpe;
  const int r0 = (bt / tilesC) << 5;
  const int c0 = (bt % tilesC) << 5;
  const float* s = src + ((size_t)e * R + r0) * C + c0;
  u16*         d = dst + ((size_t)e * C + c0) * R + r0;
  const int tx = threadIdx.x & 31, ty = threadIdx.x >> 5;
  #pragma unroll
  for (int i = 0; i < 4; ++i) {
    int r = ty + (i << 3);
    tile[r][tx] = s[(size_t)r * C + tx];          // coalesced f32 reads
  }
  __syncthreads();
  #pragma unroll
  for (int i = 0; i < 4; ++i) {
    int r = ty + (i << 3);
    d[(size_t)r * R + tx] = f2bf(tile[tx][r]);    // coalesced bf16 writes
  }
}

__global__ __launch_bounds__(256)
void moe_wmma_kernel(const float* __restrict__ x,
                     const float* __restrict__ b1, const float* __restrict__ b2,
                     const u16* __restrict__ W1T,  // [E][H][D] bf16 (N-major, K contig)
                     const u16* __restrict__ W2T,  // [E][D][H] bf16
                     const int* __restrict__ counts, const int* __restrict__ rows,
                     float* __restrict__ out)
{
  __shared__ u16 xs[16 * XS_STRIDE];   //  8.4 KB gathered x rows (bf16)
  __shared__ u16 hs[16 * HS_STRIDE];   // 33.0 KB hidden tile (bf16)
  __shared__ int gidx[16];

  const int e     = blockIdx.x >> 8;
  const int t0    = blockIdx.x & 255;
  const int cnt   = counts[e];
  const int rbase = t0 << 4;
  if (rbase >= cnt) return;            // uniform per block

  const int tid  = threadIdx.x;
  const int lane = tid & 31;
  const int w    = tid >> 5;           // wave 0..7
  const int nlo  = lane & 15;
  const int isHi = lane >> 4;
  const int aklo = isHi ? 8 : 0;       // A frag: lo lanes K{0..7,16..23}, hi {8..15,24..31}
  const int bko  = isHi ? 8 : 0;       // B frag: lo lanes K{0..7,16..23}, hi {8..15,24..31}
  const int mrow = nlo;

  if (tid < 16) {
    int r = rbase + tid;
    gidx[tid] = (r < cnt) ? rows[e * B_ + r] : -1;
  }
  __syncthreads();

  // gather x rows -> LDS bf16 (coalesced: one full row per block-iteration)
  #pragma unroll 4
  for (int j = 0; j < 16; ++j) {
    int gm = gidx[j];
    float v = (gm >= 0) ? x[(size_t)gm * D_ + tid] : 0.0f;
    xs[j * XS_STRIDE + tid] = f2bf(v);
  }
  __syncthreads();

  const f32x8 vzero = {0.f, 0.f, 0.f, 0.f, 0.f, 0.f, 0.f, 0.f};

  // ================= Layer 1: [16 x 256] @ [256 x 1024], no barriers =========
  const u16* w1p = W1T + (size_t)e * H_ * D_
                 + (size_t)((w << 7) + nlo) * D_ + bko;   // tile t adds t*16*D_
  f32x8 acc[8];
  #pragma unroll
  for (int t = 0; t < 8; ++t) acc[t] = vzero;

  for (int kk = 0; kk < D_ / 32; ++kk) {
    const int k0 = kk * 32;
    const u16* ap = &xs[mrow * XS_STRIDE + k0 + aklo];
    bf16x16 afrag = ld_frag(ap, ap + 16);
    bf16x16 bfrag[8];
    #pragma unroll
    for (int t = 0; t < 8; ++t) {      // one wide clause: 16 global_load_b128
      const u16* bp = w1p + (size_t)(t << 4) * D_ + k0;
      bfrag[t] = ld_frag(bp, bp + 16);
    }
    SCHED_FENCE();                     // all loads issue before any WMMA
    #pragma unroll
    for (int t = 0; t < 8; ++t)        // WMMAs drain with descending loadcnt waits
      acc[t] = __builtin_amdgcn_wmma_f32_16x16x32_bf16(
          false, afrag, false, bfrag[t], (short)0, acc[t], false, false);
  }

  // bias + ReLU -> hs (bf16)
  const float* b1e = b1 + (size_t)e * H_;
  {
    const int mb = isHi << 3;
    #pragma unroll
    for (int t = 0; t < 8; ++t) {
      int h = (w << 7) + (t << 4) + nlo;
      float bias = b1e[h];
      #pragma unroll
      for (int r = 0; r < 8; ++r) {
        float v = acc[t][r] + bias;
        v = v > 0.f ? v : 0.f;
        hs[(mb + r) * HS_STRIDE + h] = f2bf(v);
      }
    }
  }
  __syncthreads();

  // ================= Layer 2: [16 x 1024] @ [1024 x 256], no barriers ========
  const u16* w2p = W2T + (size_t)e * D_ * H_
                 + (size_t)((w << 5) + nlo) * H_ + bko;   // tile t adds t*16*H_
  f32x8 acc2[2] = {vzero, vzero};

  for (int kk = 0; kk < H_ / 32; kk += 2) {
    const int k0 = kk * 32;
    const u16* ap0 = &hs[mrow * HS_STRIDE + k0 + aklo];
    bf16x16 a0 = ld_frag(ap0, ap0 + 16);
    bf16x16 a1 = ld_frag(ap0 + 32, ap0 + 48);
    bf16x16 bfrag[4];
    #pragma unroll
    for (int t = 0; t < 2; ++t) {
      const u16* bp = w2p + (size_t)(t << 4) * H_ + k0;
      bfrag[t * 2 + 0] = ld_frag(bp,      bp + 16);
      bfrag[t * 2 + 1] = ld_frag(bp + 32, bp + 48);
    }
    SCHED_FENCE();                     // 4 B-frags + 2 A-frags in flight
    #pragma unroll
    for (int t = 0; t < 2; ++t) {
      acc2[t] = __builtin_amdgcn_wmma_f32_16x16x32_bf16(
          false, a0, false, bfrag[t * 2 + 0], (short)0, acc2[t], false, false);
      acc2[t] = __builtin_amdgcn_wmma_f32_16x16x32_bf16(
          false, a1, false, bfrag[t * 2 + 1], (short)0, acc2[t], false, false);
    }
  }

  // epilogue: + b2, scatter valid rows to out
  const float* b2e = b2 + (size_t)e * D_;
  const int mb = isHi << 3;
  #pragma unroll
  for (int t = 0; t < 2; ++t) {
    int dc = (w << 5) + (t << 4) + nlo;
    float bias = b2e[dc];
    #pragma unroll
    for (int r = 0; r < 8; ++r) {
      int gm = gidx[mb + r];
      if (gm >= 0) out[(size_t)gm * D_ + dc] = acc2[t][r] + bias;
    }
  }
}

extern "C" void kernel_launch(void* const* d_in, const int* in_sizes, int n_in,
                              void* d_out, int out_size, void* d_ws, size_t ws_size,
                              hipStream_t stream) {
  (void)in_sizes; (void)n_in; (void)out_size; (void)ws_size;
  const float* x  = (const float*)d_in[0];
  const float* W1 = (const float*)d_in[1];
  const float* b1 = (const float*)d_in[2];
  const float* W2 = (const float*)d_in[3];
  const float* b2 = (const float*)d_in[4];
  const int* eidx = (const int*)d_in[5];
  float* out = (float*)d_out;

  // workspace layout (256B aligned):
  //   [0)        counts   : E ints
  //   [256)      rows     : E*B ints              = 131072 B
  //   [131328)   W1T bf16 : E*H*D                 = 4 MB
  //   [4325632)  W2T bf16 : E*D*H                 = 4 MB
  int* counts = (int*)d_ws;
  int* rows   = (int*)((char*)d_ws + 256);
  u16* W1T    = (u16*)((char*)d_ws + 131328);
  u16* W2T    = (u16*)((char*)d_ws + 131328 + (size_t)E_ * H_ * D_ * 2);

  k_zero<<<1, 32, 0, stream>>>(counts);
  k_scatter<<<B_ / 256, 256, 0, stream>>>(eidx, counts, rows);
  // W1 [E][D][H] -> W1T [E][H][D] ; W2 [E][H][D] -> W2T [E][D][H]
  k_transpose_cvt<<<E_ * (D_ / 32) * (H_ / 32), 256, 0, stream>>>(W1, W1T, D_, H_);
  k_transpose_cvt<<<E_ * (H_ / 32) * (D_ / 32), 256, 0, stream>>>(W2, W2T, H_, D_);
  moe_wmma_kernel<<<E_ * (B_ / 16), 256, 0, stream>>>(
      x, b1, b2, W1T, W2T, counts, rows, out);
}
